// GAT_67224828117555
// MI455X (gfx1250) — matrix operand order repaired and checked by reading, not compiled, not run on hardware
//
#include <hip/hip_runtime.h>
#include <hip/hip_bf16.h>

#define F_DIM   128
#define HEADSN  4
#define HDIM    32
#define NEG_SLOPE 0.2f
#define LDT 136   // padded LDS row stride in ushort (136*2B=272B -> conflict-free 16-lane reads)

typedef __attribute__((ext_vector_type(8)))  float  v8f;
typedef __attribute__((ext_vector_type(16))) __bf16 v16bf;

union BF16Frag { v16bf v; unsigned int u[8]; };

__device__ __forceinline__ unsigned short f2bf(float f) {
    unsigned int u = __float_as_uint(f);
    u += 0x7FFFu + ((u >> 16) & 1u);          // round-to-nearest-even
    return (unsigned short)(u >> 16);
}

__device__ __forceinline__ float atomicMaxFloat(float* addr, float value) {
    if (value >= 0.0f)
        return __int_as_float(atomicMax((int*)addr, __float_as_int(value)));
    else
        return __uint_as_float(atomicMin((unsigned int*)addr, __float_as_uint(value)));
}

// ---------------------------------------------------------------------------
// GEMM: C[M x 128] = A[M x 128] * W[128 x 128], bf16 WMMA, f32 accumulate.
// Block = 256 thr (8 waves), 64-row tile, full K=128 and Ncol=128 in LDS.
// ---------------------------------------------------------------------------
__global__ __launch_bounds__(256) void gat_gemm_wmma(
    const float* __restrict__ A, const float* __restrict__ W,
    float* __restrict__ C, int M)
{
    __shared__ unsigned short lds_a[64 * LDT];    // A tile, [row][k] bf16
    __shared__ unsigned short lds_b[128 * LDT];   // W^T,   [n][k]  bf16

    const int tid  = threadIdx.x;
    const int row0 = blockIdx.x * 64;

    // stage A tile (64x128), guarded
    for (int idx = tid; idx < 64 * 128; idx += 256) {
        int r = idx >> 7, c = idx & 127;
        int gr = row0 + r;
        float v = (gr < M) ? A[gr * F_DIM + c] : 0.0f;
        lds_a[r * LDT + c] = f2bf(v);
    }
    // stage W transposed: lds_b[n][k] = W[k][n]
    for (int idx = tid; idx < 128 * 128; idx += 256) {
        int k = idx >> 7, n = idx & 127;
        lds_b[n * LDT + k] = f2bf(W[idx]);
    }
    __syncthreads();

    const int wv   = tid >> 5;
    const int lane = tid & 31;
    const int rstripe = wv >> 1;          // 0..3 -> rows rstripe*16..+15
    const int jbase   = (wv & 1) * 4;     // 4 column tiles of 16
    const int lrow    = rstripe * 16 + (lane & 15);
    const bool lo16   = (lane < 16);

    v8f acc[4];
    #pragma unroll
    for (int j = 0; j < 4; ++j)
        #pragma unroll
        for (int r = 0; r < 8; ++r) acc[j][r] = 0.0f;

    #pragma unroll
    for (int kk = 0; kk < 4; ++kk) {
        // A fragment: lane<16 holds K {0..7,16..23}+kk*32, lane>=16 adds +8
        BF16Frag af;
        const unsigned short* ap = lds_a + lrow * LDT + kk * 32 + (lo16 ? 0 : 8);
        #pragma unroll
        for (int v = 0; v < 8; ++v) {
            int koff = (v < 4) ? (2 * v) : (8 + 2 * v);   // 0,2,4,6,16,18,20,22
            af.u[v] = *(const unsigned int*)(ap + koff);
        }
        #pragma unroll
        for (int j = 0; j < 4; ++j) {
            // B fragment: lane<16 holds K {0..7,16..23}+kk*32, lane>=16 adds +8 (col n)
            int n = (jbase + j) * 16 + (lane & 15);
            const unsigned short* bp = lds_b + n * LDT + kk * 32 + (lo16 ? 0 : 8);
            BF16Frag bf;
            #pragma unroll
            for (int v = 0; v < 8; ++v)
                bf.u[v] = *(const unsigned int*)(bp + ((v < 4) ? (2 * v) : (8 + 2 * v)));
            acc[j] = __builtin_amdgcn_wmma_f32_16x16x32_bf16(
                false, af.v, false, bf.v, (short)0, acc[j], false, false);
        }
    }

    // store: VGPR r -> M = r (lanes 0-15) / 8+r (lanes 16-31)
    #pragma unroll
    for (int j = 0; j < 4; ++j) {
        int n = (jbase + j) * 16 + (lane & 15);
        int mbase = row0 + rstripe * 16 + (lo16 ? 0 : 8);
        #pragma unroll
        for (int r = 0; r < 8; ++r) {
            int gm = mbase + r;
            if (gm < M) C[gm * F_DIM + n] = acc[j][r];
        }
    }
}

// ---------------------------------------------------------------------------
// Per-node attention dots: a_src[n,h] = <H[n,h*32:], att_src[h]>, same for dst
// ---------------------------------------------------------------------------
__global__ void gat_attn_dots(const float* __restrict__ H,
                              const float* __restrict__ att_src,
                              const float* __restrict__ att_dst,
                              float* __restrict__ a_src,
                              float* __restrict__ a_dst, int N)
{
    int t = blockIdx.x * blockDim.x + threadIdx.x;
    if (t >= N * HEADSN) return;
    int n = t >> 2, h = t & 3;
    const float* hp = H + n * F_DIM + h * HDIM;
    const float* as = att_src + h * HDIM;
    const float* ad = att_dst + h * HDIM;
    float sa = 0.0f, sd = 0.0f;
    #pragma unroll
    for (int d = 0; d < HDIM; ++d) {
        float hv = hp[d];
        sa += hv * as[d];
        sd += hv * ad[d];
    }
    a_src[t] = sa;
    a_dst[t] = sd;
}

// ---------------------------------------------------------------------------
// Init per-layer accumulators: m=-inf, s=0, acc=0
// ---------------------------------------------------------------------------
__global__ void gat_init(float* __restrict__ m, float* __restrict__ s,
                         float* __restrict__ acc, int n4, int n128)
{
    int i = blockIdx.x * blockDim.x + threadIdx.x;
    if (i < n4) { m[i] = -__int_as_float(0x7f800000); s[i] = 0.0f; }
    if (i < n128) acc[i] = 0.0f;
}

// ---------------------------------------------------------------------------
// Pass A: segment max of leaky_relu(a_src[src]+a_dst[dst]) per (dst, head)
// Edge ids [0,E) come from edge_index; [E, E+N) are self loops.
// ---------------------------------------------------------------------------
__global__ void gat_edge_max(const int* __restrict__ ei, int E, int N,
                             const float* __restrict__ a_src,
                             const float* __restrict__ a_dst,
                             float* __restrict__ m)
{
    int e = blockIdx.x * blockDim.x + threadIdx.x;
    int Etot = E + N;
    if (e >= Etot) return;
    int s, d;
    if (e < E) {
        s = ei[e]; d = ei[E + e];
        s = (s < 0) ? 0 : (s >= N ? N - 1 : s);
        d = (d < 0) ? 0 : (d >= N ? N - 1 : d);
    } else {
        s = d = e - E;
    }
    const float4 as = *(const float4*)(a_src + s * 4);
    const float4 ad = *(const float4*)(a_dst + d * 4);
    float vs[4] = { as.x + ad.x, as.y + ad.y, as.z + ad.z, as.w + ad.w };
    #pragma unroll
    for (int h = 0; h < 4; ++h) {
        float v = vs[h];
        v = (v > 0.0f) ? v : NEG_SLOPE * v;
        atomicMaxFloat(m + d * 4 + h, v);
    }
}

// ---------------------------------------------------------------------------
// Pass B: one wave per edge. w = exp(lrelu(...) - m[dst]);
// s[dst,h] += w; acc[dst,:] += w * H[src,:]
// ---------------------------------------------------------------------------
__global__ void gat_edge_acc(const int* __restrict__ ei, int E, int N,
                             const float* __restrict__ H,
                             const float* __restrict__ a_src,
                             const float* __restrict__ a_dst,
                             const float* __restrict__ m,
                             float* __restrict__ ssum,
                             float* __restrict__ acc)
{
    int wid  = (blockIdx.x * blockDim.x + threadIdx.x) >> 5;
    int lane = threadIdx.x & 31;
    int Etot = E + N;
    if (wid >= Etot) return;
    int s, d;
    if (wid < E) {
        s = ei[wid]; d = ei[E + wid];
        s = (s < 0) ? 0 : (s >= N ? N - 1 : s);
        d = (d < 0) ? 0 : (d >= N ? N - 1 : d);
    } else {
        s = d = wid - E;
    }
    int head = lane >> 3;
    float ev = a_src[s * 4 + head] + a_dst[d * 4 + head];
    ev = (ev > 0.0f) ? ev : NEG_SLOPE * ev;
    float w = __expf(ev - m[d * 4 + head]);
    if ((lane & 7) == 0) atomicAdd(ssum + d * 4 + head, w);
    const float4 hv = *(const float4*)(H + s * F_DIM + lane * 4);
    float* ao = acc + d * F_DIM + lane * 4;
    atomicAdd(ao + 0, w * hv.x);
    atomicAdd(ao + 1, w * hv.y);
    atomicAdd(ao + 2, w * hv.z);
    atomicAdd(ao + 3, w * hv.w);
}

// ---------------------------------------------------------------------------
// Finalize: out = relu(acc / (s + 1e-16) + bias), in place.
// ---------------------------------------------------------------------------
__global__ void gat_finalize(float* __restrict__ acc,
                             const float* __restrict__ ssum,
                             const float* __restrict__ bias, int N)
{
    int i = blockIdx.x * blockDim.x + threadIdx.x;
    if (i >= N * F_DIM) return;
    int n = i >> 7, dd = i & 127;
    float v = acc[i] / (ssum[n * 4 + (dd >> 5)] + 1e-16f) + bias[dd];
    acc[i] = (v > 0.0f) ? v : 0.0f;
}

// ---------------------------------------------------------------------------
// Classifier: out[N x 40] = H[N x 128] @ Wc[128 x 40] + bc, 16 nodes / block
// ---------------------------------------------------------------------------
__global__ __launch_bounds__(256) void gat_classifier(
    const float* __restrict__ H, const float* __restrict__ Wc,
    const float* __restrict__ bc, float* __restrict__ out, int N, int Cn)
{
    __shared__ float wcs[128 * 40];
    __shared__ float hs[16 * 128];
    int tid = threadIdx.x;
    for (int i = tid; i < 128 * Cn; i += 256) wcs[i] = Wc[i];
    int n0 = blockIdx.x * 16;
    for (int i = tid; i < 16 * 128; i += 256) {
        int r = i >> 7, gn = n0 + r;
        hs[i] = (gn < N) ? H[gn * F_DIM + (i & 127)] : 0.0f;
    }
    __syncthreads();
    for (int oi = tid; oi < 16 * Cn; oi += 256) {
        int r = oi / Cn, c = oi % Cn;
        int gn = n0 + r;
        if (gn >= N) continue;
        float a = bc[c];
        const float* hrow = hs + r * 128;
        #pragma unroll 8
        for (int k = 0; k < 128; ++k) a += hrow[k] * wcs[k * Cn + c];
        out[gn * Cn + c] = a;
    }
}

// ---------------------------------------------------------------------------
extern "C" void kernel_launch(void* const* d_in, const int* in_sizes, int n_in,
                              void* d_out, int out_size, void* d_ws, size_t ws_size,
                              hipStream_t stream)
{
    const float* x    = (const float*)d_in[0];
    const int*   ei   = (const int*)  d_in[1];
    const float* W1   = (const float*)d_in[2];
    const float* as1  = (const float*)d_in[3];
    const float* ad1  = (const float*)d_in[4];
    const float* b1   = (const float*)d_in[5];
    const float* W2   = (const float*)d_in[6];
    const float* as2  = (const float*)d_in[7];
    const float* ad2  = (const float*)d_in[8];
    const float* b2   = (const float*)d_in[9];
    const float* Wc   = (const float*)d_in[10];
    const float* bc   = (const float*)d_in[11];
    float* out = (float*)d_out;

    const int N = in_sizes[0] / F_DIM;     // 50000
    const int E = in_sizes[1] / 2;         // 800000
    const int Cn = in_sizes[11];           // 40
    const int Etot = E + N;

    float* ws = (float*)d_ws;
    float* Hpre  = ws;                          // N*128
    float* Agg   = Hpre + (size_t)N * F_DIM;    // N*128
    float* a_src = Agg  + (size_t)N * F_DIM;    // N*4
    float* a_dst = a_src + (size_t)N * 4;       // N*4
    float* mmax  = a_dst + (size_t)N * 4;       // N*4
    float* ssum  = mmax + (size_t)N * 4;        // N*4

    const int gGemm = (N + 63) / 64;
    const int gAttn = (N * HEADSN + 255) / 256;
    const int gInit = (N * F_DIM + 255) / 256;
    const int gEmax = (Etot + 255) / 256;
    const int gEacc = (Etot + 7) / 8;           // 8 waves per 256-thread block
    const int gFin  = (N * F_DIM + 255) / 256;
    const int gCls  = (N + 15) / 16;

    // ---- Layer 1 ----
    gat_gemm_wmma<<<gGemm, 256, 0, stream>>>(x, W1, Hpre, N);
    gat_attn_dots<<<gAttn, 256, 0, stream>>>(Hpre, as1, ad1, a_src, a_dst, N);
    gat_init<<<gInit, 256, 0, stream>>>(mmax, ssum, Agg, N * 4, N * F_DIM);
    gat_edge_max<<<gEmax, 256, 0, stream>>>(ei, E, N, a_src, a_dst, mmax);
    gat_edge_acc<<<gEacc, 256, 0, stream>>>(ei, E, N, Hpre, a_src, a_dst, mmax, ssum, Agg);
    gat_finalize<<<gFin, 256, 0, stream>>>(Agg, ssum, b1, N);

    // ---- Layer 2 (Agg holds layer-1 output; safe to overwrite after GEMM) ----
    gat_gemm_wmma<<<gGemm, 256, 0, stream>>>(Agg, W2, Hpre, N);
    gat_attn_dots<<<gAttn, 256, 0, stream>>>(Hpre, as2, ad2, a_src, a_dst, N);
    gat_init<<<gInit, 256, 0, stream>>>(mmax, ssum, Agg, N * 4, N * F_DIM);
    gat_edge_max<<<gEmax, 256, 0, stream>>>(ei, E, N, a_src, a_dst, mmax);
    gat_edge_acc<<<gEacc, 256, 0, stream>>>(ei, E, N, Hpre, a_src, a_dst, mmax, ssum, Agg);
    gat_finalize<<<gFin, 256, 0, stream>>>(Agg, ssum, b2, N);

    // ---- Classifier ----
    gat_classifier<<<gCls, 256, 0, stream>>>(Agg, Wc, bc, out, N, Cn);
}
